// MIMIDP_40716289966458
// MI455X (gfx1250) — compile-verified, matrix operand order checked by reading, not run on hardware
//
#include <hip/hip_runtime.h>
#include <hip/hip_bf16.h>
#include <math.h>

// ---------------- types ----------------
typedef __attribute__((ext_vector_type(16))) __bf16 v16bf;
typedef __attribute__((ext_vector_type(8)))  float  v8f;
typedef unsigned short ushort_t;

union ABv { v16bf v; uint4 q[2]; };

static __device__ __forceinline__ ushort_t f2bf(float f) {
    unsigned u = __float_as_uint(f);
    unsigned r = (u + 0x7FFFu + ((u >> 16) & 1u)) >> 16;   // RNE
    return (ushort_t)r;
}

static __device__ __forceinline__ v8f wmma_bf(const ABv& a, const ABv& b, v8f c) {
    return __builtin_amdgcn_wmma_f32_16x16x32_bf16(
               false, a.v, false, b.v, (short)0, c, false, false);
}

// ---------------- WMMA NT GEMM, K == 64 fixed -------------------------------
// C[M,N] = A[M,64] * B[N,64]^T (bf16 in, f32 out).
// Grid: blockIdx.y = m-tile (16 rows), blockIdx.x covers n-groups of 4 tiles
// (16x64 strip per wave). A tile is loaded into VGPRs once and reused across
// the 4 WMMA n-tiles. Tail guard on n-tiles is wave-uniform (EXEC stays ~0
// through every WMMA, as the ISA requires).
__global__ void gemm_nt_bf16_k64(const ushort_t* __restrict__ A,
                                 const ushort_t* __restrict__ Bm,
                                 float* __restrict__ C, int N) {
    int wpb  = blockDim.x >> 5;
    int ng   = blockIdx.x * wpb + (threadIdx.x >> 5);   // n-group (4 tiles of 16)
    int lane = threadIdx.x & 31;
    int ntN  = N >> 4;
    if (ng > ((ntN - 1) >> 2)) return;
    int mt   = blockIdx.y;
    int half = lane >> 4;                 // 0: K 0..7 & 16..23 ; 1: K 8..15 & 24..31
    int l15  = lane & 15;

    // A tile: 16x64, resident in VGPRs for all 4 n-tiles
    const ushort_t* pa = A + (size_t)((mt << 4) + l15) * 64 + half * 8;
    ABv a0, a1;
    a0.q[0] = *(const uint4*)(pa);        a0.q[1] = *(const uint4*)(pa + 16);
    a1.q[0] = *(const uint4*)(pa + 32);   a1.q[1] = *(const uint4*)(pa + 48);

    int m0 = (mt << 4) + (half << 3);
#pragma unroll
    for (int s = 0; s < 4; ++s) {
        int nt = (ng << 2) + s;           // wave-uniform
        if (nt >= ntN) break;
        const ushort_t* pb = Bm + (size_t)((nt << 4) + l15) * 64 + half * 8;
        ABv b0, b1;
        b0.q[0] = *(const uint4*)(pb);        b0.q[1] = *(const uint4*)(pb + 16);
        b1.q[0] = *(const uint4*)(pb + 32);   b1.q[1] = *(const uint4*)(pb + 48);
        v8f acc = {};
        acc = wmma_bf(a0, b0, acc);
        acc = wmma_bf(a1, b1, acc);
        float* pc = C + (size_t)m0 * N + (nt << 4) + l15;
#pragma unroll
        for (int r = 0; r < 8; ++r) pc[(size_t)r * N] = acc[r];
    }
}

// ---------------- elementwise / utility kernels ----------------
__global__ void k_zero(float* p, long n) {
    long i = (long)blockIdx.x * blockDim.x + threadIdx.x;
    if (i < n) p[i] = 0.f;
}
__global__ void k_cvt_bf(const float* __restrict__ x, ushort_t* __restrict__ y, long n) {
    long i = (long)blockIdx.x * blockDim.x + threadIdx.x;
    if (i < n) y[i] = f2bf(x[i]);
}
__global__ void k_scale(float* p, float s, long n) {
    long i = (long)blockIdx.x * blockDim.x + threadIdx.x;
    if (i < n) p[i] *= s;
}
// WT[mat][n][k] = f2bf(W[mat][k][n]) for 64x64 weight matrices
__global__ void k_transpose_w(const float* __restrict__ W, ushort_t* __restrict__ WT, int nmats) {
    int i = blockIdx.x * blockDim.x + threadIdx.x;
    if (i >= nmats * 4096) return;
    int mat = i >> 12, n = (i >> 6) & 63, k = i & 63;
    WT[i] = f2bf(W[(mat << 12) + (k << 6) + n]);
}
__global__ void k_watt(const float* __restrict__ att_m, const float* __restrict__ att,
                       float* __restrict__ watt) {
    int k = threadIdx.x;
    if (k >= 64) return;
    float s = 0.f;
    for (int d = 0; d < 64; ++d) s += att_m[k * 64 + d] * att[d];
    watt[k] = s;
}
// ---------------- hypergraph conv ----------------
__global__ void k_deg(const int* __restrict__ node, const int* __restrict__ hedge,
                      float* Ddeg, float* Bdeg, int E) {
    int e = blockIdx.x * blockDim.x + threadIdx.x;
    if (e >= E) return;
    atomicAdd(&Ddeg[node[e]], 1.f);
    atomicAdd(&Bdeg[hedge[e]], 1.f);
}
__global__ void k_invdeg(const float* deg, float* inv, int n) {
    int i = blockIdx.x * blockDim.x + threadIdx.x;
    if (i < n) inv[i] = deg[i] > 0.f ? 1.f / deg[i] : 0.f;
}
__global__ void k_gate_act(const float* __restrict__ emb, const float* __restrict__ g,
                           const float* __restrict__ bias, float* cur, float* acc, long n64) {
    long i = (long)blockIdx.x * blockDim.x + threadIdx.x;
    if (i >= n64) return;
    int d = (int)(i & 63);
    float s = 1.f / (1.f + __expf(-(g[i] + bias[d])));
    float v = emb[i] * s;
    cur[i] = v; acc[i] = v;
}
// m[hedge[e]][d] += xt[node[e]][d]
__global__ void k_scatter1(const int* __restrict__ node, const int* __restrict__ hedge,
                           const float* __restrict__ xt, float* __restrict__ m, long E64) {
    long i = (long)blockIdx.x * blockDim.x + threadIdx.x;
    if (i >= E64) return;
    int e = (int)(i >> 6), d = (int)(i & 63);
    atomicAdd(&m[(size_t)hedge[e] * 64 + d], xt[(size_t)node[e] * 64 + d]);
}
// cur[node[e]][d] += m[hedge[e]][d] * invB[hedge[e]]
__global__ void k_scatter2(const int* __restrict__ node, const int* __restrict__ hedge,
                           const float* __restrict__ m, const float* __restrict__ invB,
                           float* __restrict__ cur, long E64) {
    long i = (long)blockIdx.x * blockDim.x + threadIdx.x;
    if (i >= E64) return;
    int e = (int)(i >> 6), d = (int)(i & 63);
    int h = hedge[e];
    atomicAdd(&cur[(size_t)node[e] * 64 + d], m[(size_t)h * 64 + d] * invB[h]);
}
__global__ void k_layer_fin(float* __restrict__ cur, const float* __restrict__ invD,
                            const float* __restrict__ bias, long n64) {
    long i = (long)blockIdx.x * blockDim.x + threadIdx.x;
    if (i >= n64) return;
    cur[i] = cur[i] * invD[i >> 6] + bias[i & 63];
}
__global__ void k_rownorm_acc(const float* __restrict__ cur, float* __restrict__ acc, int N) {
    int n = blockIdx.x * blockDim.x + threadIdx.x;
    if (n >= N) return;
    const float* r = cur + (size_t)n * 64;
    float s = 0.f;
    for (int d = 0; d < 64; ++d) { float v = r[d]; s += v * v; }
    float inv = 1.f / fmaxf(sqrtf(s), 1e-12f);
    float* a = acc + (size_t)n * 64;
    for (int d = 0; d < 64; ++d) a[d] += r[d] * inv;
}
// channel-attention softmax combine -> HG (bf16)
__global__ void k_combine(const float* __restrict__ c0, const float* __restrict__ c1,
                          const float* __restrict__ c2, const float* __restrict__ watt,
                          ushort_t* __restrict__ HGb, int N) {
    int n = blockIdx.x * blockDim.x + threadIdx.x;
    if (n >= N) return;
    const float* p0 = c0 + (size_t)n * 64;
    const float* p1 = c1 + (size_t)n * 64;
    const float* p2 = c2 + (size_t)n * 64;
    float l0 = 0.f, l1 = 0.f, l2 = 0.f;
    for (int d = 0; d < 64; ++d) {
        float w = watt[d];
        l0 += p0[d] * w; l1 += p1[d] * w; l2 += p2[d] * w;
    }
    float mx = fmaxf(l0, fmaxf(l1, l2));
    float e0 = __expf(l0 - mx), e1 = __expf(l1 - mx), e2 = __expf(l2 - mx);
    float inv = 1.f / (e0 + e1 + e2);
    float s0 = e0 * inv, s1 = e1 * inv, s2 = e2 * inv;
    ushort_t* o = HGb + (size_t)n * 64;
    for (int d = 0; d < 64; ++d) o[d] = f2bf(s0 * p0[d] + s1 * p1[d] + s2 * p2[d]);
}
// ---------------- attention path ----------------
__global__ void k_gather_bf(const ushort_t* __restrict__ HGb, const int* __restrict__ seq,
                            ushort_t* __restrict__ out, int rows) {
    int i = blockIdx.x * blockDim.x + threadIdx.x;
    if (i >= rows * 64) return;
    int r = i >> 6, d = i & 63;
    out[i] = HGb[(size_t)seq[r] * 64 + d];
}
__global__ void k_attn(const float* __restrict__ q, const float* __restrict__ k,
                       const float* __restrict__ v, const int* __restrict__ seq,
                       float* __restrict__ dist, float* __restrict__ ctx, int reverse) {
    int bh = blockIdx.x, b = bh >> 3, h = bh & 7;
    __shared__ float Qs[50][8], Ks[50][8], Vs[50][8];
    int t = threadIdx.x;
    for (int idx = t; idx < 400; idx += 64) {
        int j = idx >> 3, d = idx & 7;
        size_t off = (size_t)(b * 50 + j) * 64 + h * 8 + d;
        Qs[j][d] = q[off]; Ks[j][d] = k[off]; Vs[j][d] = v[off];
    }
    __syncthreads();
    if (t < 50) {
        int i = t;
        float sc[50], mx = -3.0e38f;
        for (int j = 0; j < 50; ++j) {
            bool allowed = reverse ? (j >= i) : (j <= i);
            float s;
            if (!allowed || seq[b * 50 + j] == 0) s = -1e9f;
            else {
                s = 0.f;
                for (int d = 0; d < 8; ++d) s += Qs[i][d] * Ks[j][d];
                s *= 0.35355339059327373f;   // 1/sqrt(8)
            }
            sc[j] = s; mx = fmaxf(mx, s);
        }
        float sum = 0.f;
        for (int j = 0; j < 50; ++j) { float e = __expf(sc[j] - mx); sc[j] = e; sum += e; }
        float inv = 1.f / sum;
        float* drow = dist + ((size_t)bh * 50 + i) * 50;
        for (int j = 0; j < 50; ++j) { sc[j] *= inv; drow[j] = sc[j]; }
        for (int d = 0; d < 8; ++d) {
            float o = 0.f;
            for (int j = 0; j < 50; ++j) o += sc[j] * Vs[j][d];
            ctx[(size_t)(b * 50 + i) * 64 + h * 8 + d] = o;
        }
    }
}
// _prev_user_mask post-pass: add -1000 at {0} U {seq[b,j]: j<=i} (dedup'd)
__global__ void k_mask(const int* __restrict__ seq, float* __restrict__ out, int NN) {
    int r = blockIdx.x * blockDim.x + threadIdx.x;
    if (r >= 800) return;
    int b = r / 50, i = r % 50;
    float* row = out + (size_t)r * NN;
    bool zeroSeen = false;
    for (int j = 0; j <= i; ++j) {
        int u = seq[b * 50 + j];
        bool dup = false;
        for (int j2 = 0; j2 < j; ++j2) if (seq[b * 50 + j2] == u) { dup = true; break; }
        if (u == 0) zeroSeen = true;
        if (!dup) row[u] += -1000.f;
    }
    if (!zeroSeen) row[0] += -1000.f;
}
// l2norm over sequence axis (axis=1 of [16,50,64])
__global__ void k_seqnorm(const float* __restrict__ x, float* __restrict__ out) {
    int t = blockIdx.x * blockDim.x + threadIdx.x;
    if (t >= 16 * 64) return;
    int b = t >> 6, d = t & 63;
    float s = 0.f;
    for (int i = 0; i < 50; ++i) { float v = x[(size_t)(b * 50 + i) * 64 + d]; s += v * v; }
    float inv = 1.f / fmaxf(sqrtf(s), 1e-12f);
    for (int i = 0; i < 50; ++i) {
        size_t o = (size_t)(b * 50 + i) * 64 + d;
        out[o] = x[o] * inv;
    }
}

// ---------------- host orchestration ----------------
static inline int cdiv(long a, long b) { return (int)((a + b - 1) / b); }

extern "C" void kernel_launch(void* const* d_in, const int* in_sizes, int n_in,
                              void* d_out, int out_size, void* d_ws, size_t ws_size,
                              hipStream_t stream) {
    const int* input_seq = (const int*)d_in[0];
    const int* label_seq = (const int*)d_in[1];
    const int* edges[3] = { (const int*)d_in[4], (const int*)d_in[5], (const int*)d_in[6] };
    const float* emb     = (const float*)d_in[7];
    const float* gate_w  = (const float*)d_in[8];
    const float* gate_b  = (const float*)d_in[9];
    const float* att     = (const float*)d_in[10];
    const float* att_m   = (const float*)d_in[11];
    const float* theta   = (const float*)d_in[12];
    const float* theta_b = (const float*)d_in[13];
    const float* w_hist  = (const float*)d_in[14];
    const float* w_fut   = (const float*)d_in[15];

    const int NN = in_sizes[7] / 64;          // 50000
    const int E  = in_sizes[4] / 2;           // 800000
    const long NN64 = (long)NN * 64;
    const long E64  = (long)E * 64;
    const int ROWS = 800;                      // B*L

    float* out = (float*)d_out;
    const long o1 = (long)ROWS * NN;           // future_output
    const long o2 = o1 * 2;                    // past_dist
    const long o3 = o2 + 320000;               // fut_dist
    const long o4 = o3 + 320000;               // norm past
    const long o5 = o4 + 51200;                // norm fut

    // bump allocator on d_ws (256B aligned)
    char* base = (char*)d_ws;
    size_t off = 0;
    auto alloc = [&](size_t bytes) -> void* {
        void* p = base + off;
        off += (bytes + 255) & ~(size_t)255;
        return p;
    };
    ushort_t* emb_bf   = (ushort_t*)alloc(NN64 * 2);
    ushort_t* cur_bf   = (ushort_t*)alloc(NN64 * 2);
    ushort_t* HG_bf    = (ushort_t*)alloc(NN64 * 2);
    ushort_t* gateT    = (ushort_t*)alloc(3 * 4096 * 2);
    ushort_t* thetaT   = (ushort_t*)alloc(2 * 4096 * 2);
    ushort_t* whistT   = (ushort_t*)alloc(4 * 4096 * 2);
    ushort_t* wfutT    = (ushort_t*)alloc(4 * 4096 * 2);
    ushort_t* x_bf     = (ushort_t*)alloc((size_t)ROWS * 64 * 2);
    ushort_t* ctx_bf   = (ushort_t*)alloc((size_t)ROWS * 64 * 2);
    ushort_t* ao_bf    = (ushort_t*)alloc((size_t)ROWS * 64 * 2);
    float* g    = (float*)alloc(NN64 * 4);
    float* xt   = (float*)alloc(NN64 * 4);
    float* mbuf = (float*)alloc(NN64 * 4);
    float* cur  = (float*)alloc(NN64 * 4);
    float* chans = (float*)alloc(3 * NN64 * 4);
    float* Bdeg = (float*)alloc((size_t)NN * 4);
    float* Ddeg = (float*)alloc((size_t)NN * 4);
    float* invB = (float*)alloc((size_t)NN * 4);
    float* invD = (float*)alloc((size_t)NN * 4);
    float* watt = (float*)alloc(64 * 4);
    float* qb   = (float*)alloc((size_t)ROWS * 64 * 4);
    float* kb   = (float*)alloc((size_t)ROWS * 64 * 4);
    float* vb   = (float*)alloc((size_t)ROWS * 64 * 4);
    float* ctx  = (float*)alloc((size_t)ROWS * 64 * 4);
    float* aout = (float*)alloc((size_t)ROWS * 64 * 4);
    (void)ws_size; (void)n_in; (void)out_size;

    // C[M,N] = A[M,64] * B[N,64]^T ; grid.y = M/16, grid.x covers n-groups of 4 tiles
    auto gemm = [&](const ushort_t* A, const ushort_t* Bm, float* C, int M, int N) {
        int nt4 = ((N >> 4) + 3) >> 2;                 // n-groups of 4 tiles
        int wpb = nt4 < 8 ? nt4 : 8;                   // waves per block
        dim3 grid(cdiv(nt4, wpb), M >> 4);
        gemm_nt_bf16_k64<<<grid, wpb * 32, 0, stream>>>(A, Bm, C, N);
    };

    // weight prep
    k_cvt_bf<<<cdiv(NN64, 256), 256, 0, stream>>>(emb, emb_bf, NN64);
    k_transpose_w<<<cdiv(3 * 4096, 256), 256, 0, stream>>>(gate_w, gateT, 3);
    k_transpose_w<<<cdiv(2 * 4096, 256), 256, 0, stream>>>(theta, thetaT, 2);
    k_transpose_w<<<cdiv(4 * 4096, 256), 256, 0, stream>>>(w_hist, whistT, 4);
    k_transpose_w<<<cdiv(4 * 4096, 256), 256, 0, stream>>>(w_fut, wfutT, 4);
    k_watt<<<1, 64, 0, stream>>>(att_m, att, watt);

    // ---- 3-channel hypergraph encoder ----
    for (int c = 0; c < 3; ++c) {
        const int* node  = edges[c];
        const int* hedge = edges[c] + E;
        float* acc = chans + (size_t)c * NN64;
        k_zero<<<cdiv(NN, 256), 256, 0, stream>>>(Bdeg, NN);
        k_zero<<<cdiv(NN, 256), 256, 0, stream>>>(Ddeg, NN);
        k_deg<<<cdiv(E, 256), 256, 0, stream>>>(node, hedge, Ddeg, Bdeg, E);
        k_invdeg<<<cdiv(NN, 256), 256, 0, stream>>>(Bdeg, invB, NN);
        k_invdeg<<<cdiv(NN, 256), 256, 0, stream>>>(Ddeg, invD, NN);
        // self-gating: x = emb * sigmoid(emb @ gate_w[c] + gate_b[c])
        gemm(emb_bf, gateT + c * 4096, g, NN, 64);
        k_gate_act<<<cdiv(NN64, 256), 256, 0, stream>>>(emb, g, gate_b + c * 64, cur, acc, NN64);
        for (int l = 0; l < 2; ++l) {
            k_cvt_bf<<<cdiv(NN64, 256), 256, 0, stream>>>(cur, cur_bf, NN64);
            gemm(cur_bf, thetaT + l * 4096, xt, NN, 64);
            k_zero<<<cdiv(NN64, 256), 256, 0, stream>>>(mbuf, NN64);
            k_scatter1<<<cdiv(E64, 256), 256, 0, stream>>>(node, hedge, xt, mbuf, E64);
            k_zero<<<cdiv(NN64, 256), 256, 0, stream>>>(cur, NN64);
            k_scatter2<<<cdiv(E64, 256), 256, 0, stream>>>(node, hedge, mbuf, invB, cur, E64);
            k_layer_fin<<<cdiv(NN64, 256), 256, 0, stream>>>(cur, invD, theta_b + l * 64, NN64);
            k_rownorm_acc<<<cdiv(NN, 256), 256, 0, stream>>>(cur, acc, NN);
        }
        k_scale<<<cdiv(NN64, 256), 256, 0, stream>>>(acc, 1.f / 3.f, NN64);
    }
    k_combine<<<cdiv(NN, 256), 256, 0, stream>>>(chans, chans + NN64, chans + 2 * NN64,
                                                 watt, HG_bf, NN);

    // ---- two attention branches ----
    for (int br = 0; br < 2; ++br) {
        const int* seq = br == 0 ? input_seq : label_seq;
        const ushort_t* wT = br == 0 ? whistT : wfutT;
        float* dist = out + (br == 0 ? o2 : o3);
        float* big  = out + (br == 0 ? 0 : o1);
        float* nrm  = out + (br == 0 ? o4 : o5);

        k_gather_bf<<<cdiv(ROWS * 64, 256), 256, 0, stream>>>(HG_bf, seq, x_bf, ROWS);
        gemm(x_bf, wT + 0 * 4096, qb, ROWS, 64);
        gemm(x_bf, wT + 1 * 4096, kb, ROWS, 64);
        gemm(x_bf, wT + 2 * 4096, vb, ROWS, 64);
        k_attn<<<128, 64, 0, stream>>>(qb, kb, vb, seq, dist, ctx, br);
        k_cvt_bf<<<cdiv(ROWS * 64, 256), 256, 0, stream>>>(ctx, ctx_bf, (long)ROWS * 64);
        gemm(ctx_bf, wT + 3 * 4096, aout, ROWS, 64);
        k_cvt_bf<<<cdiv(ROWS * 64, 256), 256, 0, stream>>>(aout, ao_bf, (long)ROWS * 64);
        // the big one: [800 x 64] @ [64 x 50000]  (HG row-major is exactly the NT B operand)
        gemm(ao_bf, HG_bf, big, ROWS, NN);
        if (br == 0) k_mask<<<cdiv(800, 256), 256, 0, stream>>>(input_seq, big, NN);
        k_seqnorm<<<cdiv(16 * 64, 256), 256, 0, stream>>>(aout, nrm);
    }
}
